// GeneSAGE_2680059593393
// MI455X (gfx1250) — compile-verified, matrix operand order checked by reading, not run on hardware
//
#include <hip/hip_runtime.h>
#include <hip/hip_bf16.h>
#include <math.h>

typedef __attribute__((ext_vector_type(16))) __bf16 v16bf;
typedef __attribute__((ext_vector_type(8)))  __bf16 v8bf;
typedef __attribute__((ext_vector_type(8)))  float  v8f;

#define N_NODES 50000
#define N_EDGES 800000
#define IN_DIM  128
#define HID     256

#define ATOMIC_ADD_F32(p, v) \
    __hip_atomic_fetch_add((p), (v), __ATOMIC_RELAXED, __HIP_MEMORY_SCOPE_AGENT)

// ---------------------------------------------------------------------------
// Zero scratch (agg, cnt, agg2 region)
__global__ void k_zero(float* __restrict__ p, int n) {
    int i = blockIdx.x * blockDim.x + threadIdx.x;
    int stride = gridDim.x * blockDim.x;
    for (; i < n; i += stride) p[i] = 0.0f;
}

// ---------------------------------------------------------------------------
// Prep: stacked B matrix [256 cols][256 K] in N-major bf16, fused bias.
//   K 0..127   -> W1_l
//   K 128..255 -> W1_r + W_skip
__global__ void k_prep(const float* __restrict__ W1_l,
                       const float* __restrict__ W1_r,
                       const float* __restrict__ W_skip,
                       const float* __restrict__ b1_l,
                       const float* __restrict__ b_skip,
                       __bf16* __restrict__ Bbuf,
                       float* __restrict__ biasc) {
    int n = threadIdx.x;  // column 0..255
    biasc[n] = b1_l[n] + b_skip[n];
    for (int k = 0; k < 256; ++k) {
        float v = (k < IN_DIM) ? W1_l[k * HID + n]
                               : (W1_r[(k - IN_DIM) * HID + n] +
                                  W_skip[(k - IN_DIM) * HID + n]);
        Bbuf[n * 256 + k] = (__bf16)v;
    }
}

// ---------------------------------------------------------------------------
// Layer-1 edge scatter: one wave per edge, lane covers 4 of 128 dims.
// agg (25.6MB) and x (25.6MB) are L2-resident -> atomics land in L2.
__global__ void k_scatter1(const int* __restrict__ ei,
                           const float* __restrict__ x,
                           float* __restrict__ agg,
                           float* __restrict__ cnt) {
    int wid  = (blockIdx.x * blockDim.x + threadIdx.x) >> 5;
    int lane = threadIdx.x & 31;
    if (wid >= N_EDGES) return;
    int src = ei[wid];
    int dst = ei[N_EDGES + wid];
    const float4 v = *(const float4*)(x + (size_t)src * IN_DIM + lane * 4);
    float* base = agg + (size_t)dst * IN_DIM + lane * 4;
    ATOMIC_ADD_F32(base + 0, v.x);
    ATOMIC_ADD_F32(base + 1, v.y);
    ATOMIC_ADD_F32(base + 2, v.z);
    ATOMIC_ADD_F32(base + 3, v.w);
    if (lane == 0) ATOMIC_ADD_F32(cnt + dst, 1.0f);
}

// ---------------------------------------------------------------------------
// Fused layer-1: [16 nodes x 256K] @ [256K x 256] via bf16 WMMA, f32 acc,
// + bias + LayerNorm + ELU, store x1 as bf16.
// Block = 256 threads = 8 waves; wave w owns columns [32w,32w+32) as two 16-col tiles.
__launch_bounds__(256, 2)
__global__ void k_layer1(const float* __restrict__ x,
                         const float* __restrict__ agg,
                         const float* __restrict__ cnt,
                         const __bf16* __restrict__ Bbuf,
                         const float* __restrict__ biasc,
                         const float* __restrict__ ln_g,
                         const float* __restrict__ ln_b,
                         __bf16* __restrict__ x1) {
    __shared__ __bf16 Atile[16][264];      // 16 rows x 256 K (+pad)
    __shared__ float  Ctile[16][260];      // 16 rows x 256 cols (+pad)
    __shared__ float  mu_s[16], rs_s[16];

    const int nodeBase = blockIdx.x * 16;
    const int t = threadIdx.x;

    // ---- stage A tile: K[0:128)=mean aggregation, K[128:256)=x ----
    {
        int r = t >> 4;              // row 0..15
        int chunk = t & 15;          // 16 elems per thread
        int node = nodeBase + r;
        float inv = 1.0f / fmaxf(cnt[node], 1.0f);
        if (chunk < 8) {
            int k0 = chunk * 16;
            const float4* ap = (const float4*)(agg + (size_t)node * IN_DIM + k0);
            #pragma unroll
            for (int q = 0; q < 4; ++q) {
                float4 v = ap[q];
                Atile[r][k0 + q * 4 + 0] = (__bf16)(v.x * inv);
                Atile[r][k0 + q * 4 + 1] = (__bf16)(v.y * inv);
                Atile[r][k0 + q * 4 + 2] = (__bf16)(v.z * inv);
                Atile[r][k0 + q * 4 + 3] = (__bf16)(v.w * inv);
            }
        } else {
            int k0 = (chunk - 8) * 16;
            const float4* xp = (const float4*)(x + (size_t)node * IN_DIM + k0);
            #pragma unroll
            for (int q = 0; q < 4; ++q) {
                float4 v = xp[q];
                Atile[r][128 + k0 + q * 4 + 0] = (__bf16)v.x;
                Atile[r][128 + k0 + q * 4 + 1] = (__bf16)v.y;
                Atile[r][128 + k0 + q * 4 + 2] = (__bf16)v.z;
                Atile[r][128 + k0 + q * 4 + 3] = (__bf16)v.w;
            }
        }
    }
    __syncthreads();

    // ---- WMMA: 8 K-steps x 2 column tiles per wave ----
    const int w     = t >> 5;
    const int lane  = t & 31;
    const int row16 = lane & 15;
    const int koff  = (lane >> 4) << 3;   // 0 or 8 (A/B 16-bit lane layout)
    const int n0    = w * 32;

    v8f acc0 = {0.f, 0.f, 0.f, 0.f, 0.f, 0.f, 0.f, 0.f};
    v8f acc1 = {0.f, 0.f, 0.f, 0.f, 0.f, 0.f, 0.f, 0.f};

    #pragma unroll
    for (int kk = 0; kk < 8; ++kk) {
        int k0 = kk * 32;
        v8bf a_lo = *(const v8bf*)&Atile[row16][k0 + koff];
        v8bf a_hi = *(const v8bf*)&Atile[row16][k0 + 16 + koff];
        v16bf a = __builtin_shufflevector(a_lo, a_hi,
                      0, 1, 2, 3, 4, 5, 6, 7, 8, 9, 10, 11, 12, 13, 14, 15);

        const __bf16* bp0 = Bbuf + (size_t)(n0 + row16) * 256 + k0 + koff;
        const __bf16* bp1 = Bbuf + (size_t)(n0 + 16 + row16) * 256 + k0 + koff;
        v16bf b0 = __builtin_shufflevector(*(const v8bf*)bp0, *(const v8bf*)(bp0 + 16),
                      0, 1, 2, 3, 4, 5, 6, 7, 8, 9, 10, 11, 12, 13, 14, 15);
        v16bf b1 = __builtin_shufflevector(*(const v8bf*)bp1, *(const v8bf*)(bp1 + 16),
                      0, 1, 2, 3, 4, 5, 6, 7, 8, 9, 10, 11, 12, 13, 14, 15);

        acc0 = __builtin_amdgcn_wmma_f32_16x16x32_bf16(
                   false, a, false, b0, (short)0, acc0, false, false);
        acc1 = __builtin_amdgcn_wmma_f32_16x16x32_bf16(
                   false, a, false, b1, (short)0, acc1, false, false);
    }

    // ---- spill C (+bias) to LDS: VGPR r -> M = r + 8*(lane>=16), N = lane%16
    {
        int mofs = (lane >> 4) << 3;
        #pragma unroll
        for (int r = 0; r < 8; ++r) {
            int m  = r + mofs;
            int c0 = n0 + row16;
            int c1 = n0 + 16 + row16;
            Ctile[m][c0] = acc0[r] + biasc[c0];
            Ctile[m][c1] = acc1[r] + biasc[c1];
        }
    }
    __syncthreads();

    // ---- LayerNorm stats: one thread per row ----
    if (t < 16) {
        float s = 0.f;
        for (int c = 0; c < 256; ++c) s += Ctile[t][c];
        float mu = s * (1.0f / 256.0f);
        float v = 0.f;
        for (int c = 0; c < 256; ++c) { float d = Ctile[t][c] - mu; v += d * d; }
        mu_s[t] = mu;
        rs_s[t] = rsqrtf(v * (1.0f / 256.0f) + 1e-5f);
    }
    __syncthreads();

    // ---- apply LN + ELU, store x1 (bf16) ----
    {
        int r  = t >> 4;
        int cb = (t & 15) * 16;
        float mu = mu_s[r], rs = rs_s[r];
        size_t obase = (size_t)(nodeBase + r) * HID;
        #pragma unroll
        for (int j = 0; j < 16; ++j) {
            int c = cb + j;
            float v = (Ctile[r][c] - mu) * rs * ln_g[c] + ln_b[c];
            v = v > 0.f ? v : (__expf(v) - 1.0f);
            x1[obase + c] = (__bf16)v;
        }
    }
}

// ---------------------------------------------------------------------------
// Layer-2 projection first (mean is linear): y_l = x1@W2_l, y_r = x1@W2_r.
// One wave per node; lane covers 8 of 256 dims (16B coalesced), shuffle-reduce.
__global__ void k_layer2a(const __bf16* __restrict__ x1,
                          const float* __restrict__ W2_l,
                          const float* __restrict__ W2_r,
                          float* __restrict__ yl,
                          float* __restrict__ yr) {
    int wid  = (blockIdx.x * blockDim.x + threadIdx.x) >> 5;
    int lane = threadIdx.x & 31;
    if (wid >= N_NODES) return;
    const __bf16* row = x1 + (size_t)wid * HID;
    v8bf h8 = *(const v8bf*)(row + lane * 8);
    float a0 = 0.f, a1 = 0.f, r0 = 0.f, r1 = 0.f;
    #pragma unroll
    for (int j = 0; j < 8; ++j) {
        int k = lane * 8 + j;
        float h = (float)h8[j];
        a0 += h * W2_l[k * 2 + 0];
        a1 += h * W2_l[k * 2 + 1];
        r0 += h * W2_r[k * 2 + 0];
        r1 += h * W2_r[k * 2 + 1];
    }
    #pragma unroll
    for (int off = 16; off > 0; off >>= 1) {
        a0 += __shfl_down(a0, off, 32);
        a1 += __shfl_down(a1, off, 32);
        r0 += __shfl_down(r0, off, 32);
        r1 += __shfl_down(r1, off, 32);
    }
    if (lane == 0) {
        yl[wid * 2 + 0] = a0;
        yl[wid * 2 + 1] = a1;
        yr[wid * 2 + 0] = r0;
        yr[wid * 2 + 1] = r1;
    }
}

// ---------------------------------------------------------------------------
// Layer-2 edge scatter: only 2 floats/edge (projection-first trick).
__global__ void k_scatter2(const int* __restrict__ ei,
                           const float* __restrict__ yl,
                           float* __restrict__ agg2) {
    int e = blockIdx.x * blockDim.x + threadIdx.x;
    if (e >= N_EDGES) return;
    int src = ei[e];
    int dst = ei[N_EDGES + e];
    ATOMIC_ADD_F32(&agg2[dst * 2 + 0], yl[src * 2 + 0]);
    ATOMIC_ADD_F32(&agg2[dst * 2 + 1], yl[src * 2 + 1]);
}

// ---------------------------------------------------------------------------
// Finalize: out = agg2/cnt + y_r + b2_l
__global__ void k_final(const float* __restrict__ agg2,
                        const float* __restrict__ cnt,
                        const float* __restrict__ yr,
                        const float* __restrict__ b2_l,
                        float* __restrict__ out) {
    int i = blockIdx.x * blockDim.x + threadIdx.x;
    if (i >= N_NODES) return;
    float inv = 1.0f / fmaxf(cnt[i], 1.0f);
    out[i * 2 + 0] = agg2[i * 2 + 0] * inv + yr[i * 2 + 0] + b2_l[0];
    out[i * 2 + 1] = agg2[i * 2 + 1] * inv + yr[i * 2 + 1] + b2_l[1];
}

// ---------------------------------------------------------------------------
extern "C" void kernel_launch(void* const* d_in, const int* in_sizes, int n_in,
                              void* d_out, int out_size, void* d_ws, size_t ws_size,
                              hipStream_t stream) {
    (void)in_sizes; (void)n_in; (void)out_size; (void)ws_size;

    const float* x      = (const float*)d_in[0];
    const int*   ei     = (const int*)  d_in[1];
    const float* W1_l   = (const float*)d_in[2];
    const float* b1_l   = (const float*)d_in[3];
    const float* W1_r   = (const float*)d_in[4];
    const float* W_skip = (const float*)d_in[5];
    const float* b_skip = (const float*)d_in[6];
    const float* ln_g   = (const float*)d_in[7];
    const float* ln_b   = (const float*)d_in[8];
    const float* W2_l   = (const float*)d_in[9];
    const float* b2_l   = (const float*)d_in[10];
    const float* W2_r   = (const float*)d_in[11];
    float* out = (float*)d_out;

    // workspace layout (256B aligned)
    char* ws = (char*)d_ws;
    size_t off = 0;
    float* agg   = (float*)(ws + off); off += (size_t)N_NODES * IN_DIM * 4;  // 25.6MB
    float* cnt   = (float*)(ws + off); off += ((size_t)N_NODES * 4 + 255) & ~255ull;
    float* agg2  = (float*)(ws + off); off += ((size_t)N_NODES * 2 * 4 + 255) & ~255ull;
    size_t zeroFloats = off / 4;       // agg + cnt + agg2 span, zeroed below
    float* yl    = (float*)(ws + off); off += ((size_t)N_NODES * 2 * 4 + 255) & ~255ull;
    float* yr    = (float*)(ws + off); off += ((size_t)N_NODES * 2 * 4 + 255) & ~255ull;
    float* biasc = (float*)(ws + off); off += 256 * 4;
    __bf16* Bbuf = (__bf16*)(ws + off); off += (size_t)256 * 256 * 2;        // 128KB
    __bf16* x1   = (__bf16*)(ws + off); off += (size_t)N_NODES * HID * 2;    // 25.6MB

    k_zero<<<2048, 256, 0, stream>>>((float*)ws, (int)zeroFloats);
    k_prep<<<1, 256, 0, stream>>>(W1_l, W1_r, W_skip, b1_l, b_skip, Bbuf, biasc);
    k_scatter1<<<N_EDGES / 8, 256, 0, stream>>>(ei, x, agg, cnt);
    k_layer1<<<N_NODES / 16, 256, 0, stream>>>(x, agg, cnt, Bbuf, biasc,
                                               ln_g, ln_b, x1);
    k_layer2a<<<N_NODES / 8, 256, 0, stream>>>(x1, W2_l, W2_r, yl, yr);
    k_scatter2<<<(N_EDGES + 255) / 256, 256, 0, stream>>>(ei, yl, agg2);
    k_final<<<(N_NODES + 255) / 256, 256, 0, stream>>>(agg2, cnt, yr, b2_l, out);
}
